// CORA_GAT_5471788335228
// MI455X (gfx1250) — matrix operand order repaired and checked by reading, not compiled, not run on hardware
//
#include <hip/hip_runtime.h>
#include <hip/hip_bf16.h>
#include <stdint.h>

typedef __bf16 bf16_t;
typedef __attribute__((ext_vector_type(16))) __bf16 v16bf;
typedef __attribute__((ext_vector_type(8)))  float  v8f;

#define NEG_SLOPE 0.2f
#define EPSV 1e-16f
#define NEG_INF (-__builtin_inff())

// ---------- helpers ----------

// native f32 -> bf16 (RNE); lowers to v_cvt_*bf16_f32 on gfx1250
__device__ __forceinline__ bf16_t f2bf(float f) { return (bf16_t)f; }

// float atomic max via sign-split int/uint atomics
// (lowers to global_atomic_max_i32 / global_atomic_min_u32; init with -inf bits)
__device__ __forceinline__ void atomicMaxFloat(float* addr, float v) {
  if (v >= 0.0f) atomicMax((int*)addr, __float_as_int(v));
  else           atomicMin((unsigned int*)addr, __float_as_uint(v));
}

__device__ __forceinline__ void fetch_edge(const long long* __restrict__ ei,
                                           long long E, long long e, int& s, int& d) {
  if (e < E) { s = (int)ei[e]; d = (int)ei[E + e]; }
  else       { s = d = (int)(e - E); }        // implicit self-loop
}

// ---------- weight prep: transpose + f32->bf16 ----------
// W1 [512,64] -> W1t [64][512] bf16 ; W2 [64,16] -> W2t [16][64] bf16
__global__ void prep_weights(const float* __restrict__ W1, const float* __restrict__ W2,
                             bf16_t* __restrict__ W1t, bf16_t* __restrict__ W2t) {
  int gid = blockIdx.x * blockDim.x + threadIdx.x;
  if (gid < 64 * 512) {
    int n = gid >> 9, k = gid & 511;
    W1t[gid] = f2bf(W1[(size_t)k * 64 + n]);
  } else if (gid < 64 * 512 + 16 * 64) {
    int g2 = gid - 64 * 512;
    int n = g2 >> 6, k = g2 & 63;
    W2t[g2] = f2bf(W2[(size_t)k * 16 + n]);
  }
}

// ---------- GEMM1: h1[N,64] = x[N,512] @ W1, bf16 WMMA ----------
// block = 256 thr = 8 waves; wave computes 16 rows x 64 cols.
// B operand (W1t, col-major) staged in 64KB LDS, 32B-granule XOR swizzle.
__global__ __launch_bounds__(256) void gemm1_wmma(const float* __restrict__ x,
    const bf16_t* __restrict__ W1t, float* __restrict__ h1, int N) {
  __shared__ bf16_t Bs[64 * 512];             // 64 cols x 512 k, bf16 = 64 KB
  int t = threadIdx.x;
  for (int idx = t; idx < 4096; idx += 256) { // 4096 x 16B chunks
    int col = idx >> 6;
    int ch  = idx & 63;                       // 8 bf16 per chunk
    int g = ch >> 1, half = ch & 1;
    int gs = g ^ (col & 31);                  // swizzle 32B granules
    float4 v = *(const float4*)(W1t + (size_t)col * 512 + ch * 8);
    *(float4*)(Bs + col * 512 + gs * 16 + half * 8) = v;
  }
  __syncthreads();

  int wave = t >> 5, lane = t & 31;
  int lr = lane & 15, hi = lane >> 4;
  int rowTile = blockIdx.x * 128 + wave * 16;
  int arow = rowTile + lr; if (arow > N - 1) arow = N - 1;
  const float* xrow = x + (size_t)arow * 512;

  v8f acc[4] = {{}, {}, {}, {}};
  for (int kk = 0; kk < 16; ++kk) {
    int k0 = kk * 32;
    // A fragment: lanes 0-15 hold K 0..15, lanes 16-31 hold K 16..31
    int ka = k0 + hi * 16;
    float4 f0 = *(const float4*)(xrow + ka);
    float4 f1 = *(const float4*)(xrow + ka + 4);
    float4 f2 = *(const float4*)(xrow + ka + 8);
    float4 f3 = *(const float4*)(xrow + ka + 12);
    v16bf a;
    a[0]=f2bf(f0.x);  a[1]=f2bf(f0.y);  a[2]=f2bf(f0.z);  a[3]=f2bf(f0.w);
    a[4]=f2bf(f1.x);  a[5]=f2bf(f1.y);  a[6]=f2bf(f1.z);  a[7]=f2bf(f1.w);
    a[8]=f2bf(f2.x);  a[9]=f2bf(f2.y);  a[10]=f2bf(f2.z); a[11]=f2bf(f2.w);
    a[12]=f2bf(f3.x); a[13]=f2bf(f3.y); a[14]=f2bf(f3.z); a[15]=f2bf(f3.w);
    // B fragment: lanes 0-15 col=lr K 0..15, lanes 16-31 col=lr K 16..31
    int kb = k0 + hi * 16;
    int g = kb >> 4;
#pragma unroll
    for (int nb = 0; nb < 4; ++nb) {
      int col = nb * 16 + lr;
      const bf16_t* bp = Bs + col * 512 + ((g ^ (col & 31)) << 4);
      union { float4 q[2]; v16bf v; } bu;
      bu.q[0] = *(const float4*)bp;
      bu.q[1] = *(const float4*)(bp + 8);
      acc[nb] = __builtin_amdgcn_wmma_f32_16x16x32_bf16(
          false, a, false, bu.v, (short)0, acc[nb], false, false);
    }
  }
  if (rowTile + 16 <= N) {                    // fast path: whole tile in range
#pragma unroll
    for (int nb = 0; nb < 4; ++nb) {
      int col = nb * 16 + lr;
#pragma unroll
      for (int r = 0; r < 8; ++r) {
        int row = rowTile + hi * 8 + r;
        h1[(size_t)row * 64 + col] = acc[nb][r];
      }
    }
  } else {
#pragma unroll
    for (int nb = 0; nb < 4; ++nb) {
      int col = nb * 16 + lr;
#pragma unroll
      for (int r = 0; r < 8; ++r) {
        int row = rowTile + hi * 8 + r;
        if (row < N) h1[(size_t)row * 64 + col] = acc[nb][r];
      }
    }
  }
}

// ---------- GEMM2: h2[N,16] = hin[N,64] @ W2, bf16 WMMA ----------
__global__ __launch_bounds__(256) void gemm2_wmma(const float* __restrict__ hin,
    const bf16_t* __restrict__ W2t, float* __restrict__ h2, int N) {
  __shared__ bf16_t Bs[16 * 72];              // padded stride 72
  int t = threadIdx.x;
  for (int idx = t; idx < 128; idx += 256) {
    int col = idx >> 3, ch = idx & 7;
    float4 v = *(const float4*)(W2t + (size_t)col * 64 + ch * 8);
    *(float4*)(Bs + col * 72 + ch * 8) = v;
  }
  __syncthreads();

  int wave = t >> 5, lane = t & 31;
  int lr = lane & 15, hi = lane >> 4;
  int rowTile = blockIdx.x * 128 + wave * 16;
  int arow = rowTile + lr; if (arow > N - 1) arow = N - 1;
  const float* xr = hin + (size_t)arow * 64;

  v8f acc = {};
#pragma unroll
  for (int kk = 0; kk < 2; ++kk) {
    int k0 = kk * 32;
    int ka = k0 + hi * 16;
    float4 f0 = *(const float4*)(xr + ka);
    float4 f1 = *(const float4*)(xr + ka + 4);
    float4 f2 = *(const float4*)(xr + ka + 8);
    float4 f3 = *(const float4*)(xr + ka + 12);
    v16bf a;
    a[0]=f2bf(f0.x);  a[1]=f2bf(f0.y);  a[2]=f2bf(f0.z);  a[3]=f2bf(f0.w);
    a[4]=f2bf(f1.x);  a[5]=f2bf(f1.y);  a[6]=f2bf(f1.z);  a[7]=f2bf(f1.w);
    a[8]=f2bf(f2.x);  a[9]=f2bf(f2.y);  a[10]=f2bf(f2.z); a[11]=f2bf(f2.w);
    a[12]=f2bf(f3.x); a[13]=f2bf(f3.y); a[14]=f2bf(f3.z); a[15]=f2bf(f3.w);
    int kb = k0 + hi * 16;
    const bf16_t* bp = Bs + lr * 72 + kb;
    union { float4 q[2]; v16bf v; } bu;
    bu.q[0] = *(const float4*)bp;
    bu.q[1] = *(const float4*)(bp + 8);
    acc = __builtin_amdgcn_wmma_f32_16x16x32_bf16(
        false, a, false, bu.v, (short)0, acc, false, false);
  }
  if (rowTile + 16 <= N) {
#pragma unroll
    for (int r = 0; r < 8; ++r) {
      int row = rowTile + hi * 8 + r;
      h2[(size_t)row * 16 + lr] = acc[r];
    }
  } else {
#pragma unroll
    for (int r = 0; r < 8; ++r) {
      int row = rowTile + hi * 8 + r;
      if (row < N) h2[(size_t)row * 16 + lr] = acc[r];
    }
  }
}

// ---------- layer-1 attention prep: logits + init ----------
__global__ void att1_init(const float* __restrict__ h1,
    const float* __restrict__ att_src, const float* __restrict__ att_dst,
    float* __restrict__ asrc, float* __restrict__ adst,
    float* __restrict__ m, float* __restrict__ s, float* __restrict__ out1, int N) {
  long long gid = (long long)blockIdx.x * blockDim.x + threadIdx.x;
  if (gid >= (long long)N * 8) return;
  int i = (int)(gid >> 3), hd = (int)(gid & 7);
  const float* hr = h1 + (size_t)i * 64 + hd * 8;
  const float* as = att_src + hd * 8;
  const float* ad = att_dst + hd * 8;
  float sa = 0.f, sd = 0.f;
#pragma unroll
  for (int c = 0; c < 8; ++c) { sa += hr[c] * as[c]; sd += hr[c] * ad[c]; }
  asrc[gid] = sa; adst[gid] = sd;
  m[gid] = NEG_INF; s[gid] = 0.f;
  float* o = out1 + (size_t)i * 64 + hd * 8;
#pragma unroll
  for (int c = 0; c < 8; ++c) o[c] = 0.f;
}

// ---------- layer-1 edge passes (8 heads) ----------
__global__ void edge_max_h8(const long long* __restrict__ ei, long long E, int N,
    const float* __restrict__ asrc, const float* __restrict__ adst, float* __restrict__ m) {
  long long gid = (long long)blockIdx.x * blockDim.x + threadIdx.x;
  if (gid >= (E + N) * 8) return;
  long long e = gid >> 3; int hd = (int)(gid & 7);
  int sI, dI; fetch_edge(ei, E, e, sI, dI);
  float a = asrc[(size_t)sI * 8 + hd] + adst[(size_t)dI * 8 + hd];
  a = a > 0.f ? a : NEG_SLOPE * a;
  atomicMaxFloat(m + (size_t)dI * 8 + hd, a);
}

__global__ void edge_sum_h8(const long long* __restrict__ ei, long long E, int N,
    const float* __restrict__ asrc, const float* __restrict__ adst,
    const float* __restrict__ m, float* __restrict__ s) {
  long long gid = (long long)blockIdx.x * blockDim.x + threadIdx.x;
  if (gid >= (E + N) * 8) return;
  long long e = gid >> 3; int hd = (int)(gid & 7);
  int sI, dI; fetch_edge(ei, E, e, sI, dI);
  float a = asrc[(size_t)sI * 8 + hd] + adst[(size_t)dI * 8 + hd];
  a = a > 0.f ? a : NEG_SLOPE * a;
  float ex = __expf(a - m[(size_t)dI * 8 + hd]);
  atomicAdd(s + (size_t)dI * 8 + hd, ex);
}

__global__ void edge_agg_h8(const long long* __restrict__ ei, long long E, int N,
    const float* __restrict__ asrc, const float* __restrict__ adst,
    const float* __restrict__ m, const float* __restrict__ h1, float* __restrict__ out1) {
  long long gid = (long long)blockIdx.x * blockDim.x + threadIdx.x;
  if (gid >= (E + N) * 16) return;
  long long e = gid >> 4; int q = (int)(gid & 15);
  int c0 = q * 4, hd = c0 >> 3;
  int sI, dI; fetch_edge(ei, E, e, sI, dI);
  float a = asrc[(size_t)sI * 8 + hd] + adst[(size_t)dI * 8 + hd];
  a = a > 0.f ? a : NEG_SLOPE * a;
  float w = __expf(a - m[(size_t)dI * 8 + hd]);
  float4 hv = *(const float4*)(h1 + (size_t)sI * 64 + c0);
  float* o = out1 + (size_t)dI * 64 + c0;
  atomicAdd(o + 0, hv.x * w);
  atomicAdd(o + 1, hv.y * w);
  atomicAdd(o + 2, hv.z * w);
  atomicAdd(o + 3, hv.w * w);
}

// normalize + bias + ELU (in place)
__global__ void finalize1(float* __restrict__ out1, const float* __restrict__ s,
                          const float* __restrict__ bias, int N) {
  long long gid = (long long)blockIdx.x * blockDim.x + threadIdx.x;
  if (gid >= (long long)N * 64) return;
  int i = (int)(gid >> 6), c = (int)(gid & 63), hd = c >> 3;
  float v = out1[gid] / (s[(size_t)i * 8 + hd] + EPSV) + bias[c];
  out1[gid] = v > 0.f ? v : __expf(v) - 1.f;
}

// ---------- layer-2 attention prep ----------
__global__ void att2_init(const float* __restrict__ h2,
    const float* __restrict__ att_src, const float* __restrict__ att_dst,
    float* __restrict__ asrc, float* __restrict__ adst,
    float* __restrict__ m, float* __restrict__ s, float* __restrict__ out2, int N) {
  int i = blockIdx.x * blockDim.x + threadIdx.x;
  if (i >= N) return;
  const float* hr = h2 + (size_t)i * 16;
  float sa = 0.f, sd = 0.f;
#pragma unroll
  for (int c = 0; c < 16; ++c) { sa += hr[c] * att_src[c]; sd += hr[c] * att_dst[c]; }
  asrc[i] = sa; adst[i] = sd;
  m[i] = NEG_INF; s[i] = 0.f;
  float* o = out2 + (size_t)i * 16;
#pragma unroll
  for (int c = 0; c < 16; ++c) o[c] = 0.f;
}

__global__ void edge_max_h1(const long long* __restrict__ ei, long long E, int N,
    const float* __restrict__ asrc, const float* __restrict__ adst, float* __restrict__ m) {
  long long gid = (long long)blockIdx.x * blockDim.x + threadIdx.x;
  if (gid >= E + N) return;
  int sI, dI; fetch_edge(ei, E, gid, sI, dI);
  float a = asrc[sI] + adst[dI];
  a = a > 0.f ? a : NEG_SLOPE * a;
  atomicMaxFloat(m + dI, a);
}

__global__ void edge_sum_h1(const long long* __restrict__ ei, long long E, int N,
    const float* __restrict__ asrc, const float* __restrict__ adst,
    const float* __restrict__ m, float* __restrict__ s) {
  long long gid = (long long)blockIdx.x * blockDim.x + threadIdx.x;
  if (gid >= E + N) return;
  int sI, dI; fetch_edge(ei, E, gid, sI, dI);
  float a = asrc[sI] + adst[dI];
  a = a > 0.f ? a : NEG_SLOPE * a;
  atomicAdd(s + dI, __expf(a - m[dI]));
}

__global__ void edge_agg_h1(const long long* __restrict__ ei, long long E, int N,
    const float* __restrict__ asrc, const float* __restrict__ adst,
    const float* __restrict__ m, const float* __restrict__ h2, float* __restrict__ out2) {
  long long gid = (long long)blockIdx.x * blockDim.x + threadIdx.x;
  if (gid >= (E + N) * 4) return;
  long long e = gid >> 2; int c0 = (int)(gid & 3) * 4;
  int sI, dI; fetch_edge(ei, E, e, sI, dI);
  float a = asrc[sI] + adst[dI];
  a = a > 0.f ? a : NEG_SLOPE * a;
  float w = __expf(a - m[dI]);
  float4 hv = *(const float4*)(h2 + (size_t)sI * 16 + c0);
  float* o = out2 + (size_t)dI * 16 + c0;
  atomicAdd(o + 0, hv.x * w);
  atomicAdd(o + 1, hv.y * w);
  atomicAdd(o + 2, hv.z * w);
  atomicAdd(o + 3, hv.w * w);
}

// normalize + bias -> node_embeddings ; softmax -> out
__global__ void finalize2(const float* __restrict__ out2, const float* __restrict__ s,
                          const float* __restrict__ bias, float* __restrict__ outp, int N) {
  int i = blockIdx.x * blockDim.x + threadIdx.x;
  if (i >= N) return;
  float inv = 1.f / (s[i] + EPSV);
  float emb[16];
  float mx = NEG_INF;
#pragma unroll
  for (int c = 0; c < 16; ++c) {
    emb[c] = out2[(size_t)i * 16 + c] * inv + bias[c];
    mx = fmaxf(mx, emb[c]);
  }
  float ssum = 0.f;
  float ex[16];
#pragma unroll
  for (int c = 0; c < 16; ++c) { ex[c] = __expf(emb[c] - mx); ssum += ex[c]; }
  float rs = 1.f / ssum;
#pragma unroll
  for (int c = 0; c < 16; ++c) {
    outp[(size_t)i * 16 + c] = ex[c] * rs;                       // softmax output
    outp[(size_t)N * 16 + (size_t)i * 16 + c] = emb[c];          // node_embeddings
  }
}

// ---------- launch ----------
static inline unsigned cdivu(long long a, long long b) { return (unsigned)((a + b - 1) / b); }

extern "C" void kernel_launch(void* const* d_in, const int* in_sizes, int n_in,
                              void* d_out, int out_size, void* d_ws, size_t ws_size,
                              hipStream_t stream) {
  const float*     x        = (const float*)d_in[0];
  const long long* ei       = (const long long*)d_in[1];
  const float*     W1       = (const float*)d_in[2];
  const float*     att_src1 = (const float*)d_in[3];
  const float*     att_dst1 = (const float*)d_in[4];
  const float*     bias1    = (const float*)d_in[5];
  const float*     W2       = (const float*)d_in[6];
  const float*     att_src2 = (const float*)d_in[7];
  const float*     att_dst2 = (const float*)d_in[8];
  const float*     bias2    = (const float*)d_in[9];

  const int       N = in_sizes[0] / 512;
  const long long E = (long long)in_sizes[1] / 2;
  const long long Et = E + N;
  float* outp = (float*)d_out;

  char* p = (char*)d_ws;
  auto take = [&](size_t bytes) -> char* {
    char* r = p; p += (bytes + 255) & ~(size_t)255; return r;
  };
  bf16_t* W1t   = (bf16_t*)take(64 * 512 * sizeof(bf16_t));
  bf16_t* W2t   = (bf16_t*)take(16 * 64 * sizeof(bf16_t));
  float* h1     = (float*)take((size_t)N * 64 * 4);
  float* asrc1  = (float*)take((size_t)N * 8 * 4);
  float* adst1  = (float*)take((size_t)N * 8 * 4);
  float* m1     = (float*)take((size_t)N * 8 * 4);
  float* s1     = (float*)take((size_t)N * 8 * 4);
  float* out1   = (float*)take((size_t)N * 64 * 4);
  float* h2     = (float*)take((size_t)N * 16 * 4);
  float* asrc2  = (float*)take((size_t)N * 4);
  float* adst2  = (float*)take((size_t)N * 4);
  float* m2     = (float*)take((size_t)N * 4);
  float* s2     = (float*)take((size_t)N * 4);
  float* out2   = (float*)take((size_t)N * 16 * 4);

  const int B = 256;
  prep_weights<<<cdivu(64 * 512 + 16 * 64, B), B, 0, stream>>>(W1, W2, W1t, W2t);

  // layer 1
  gemm1_wmma<<<cdivu(N, 128), B, 0, stream>>>(x, W1t, h1, N);
  att1_init<<<cdivu((long long)N * 8, B), B, 0, stream>>>(h1, att_src1, att_dst1,
                                                          asrc1, adst1, m1, s1, out1, N);
  edge_max_h8<<<cdivu(Et * 8, B), B, 0, stream>>>(ei, E, N, asrc1, adst1, m1);
  edge_sum_h8<<<cdivu(Et * 8, B), B, 0, stream>>>(ei, E, N, asrc1, adst1, m1, s1);
  edge_agg_h8<<<cdivu(Et * 16, B), B, 0, stream>>>(ei, E, N, asrc1, adst1, m1, h1, out1);
  finalize1<<<cdivu((long long)N * 64, B), B, 0, stream>>>(out1, s1, bias1, N);

  // layer 2
  gemm2_wmma<<<cdivu(N, 128), B, 0, stream>>>(out1, W2t, h2, N);
  att2_init<<<cdivu(N, B), B, 0, stream>>>(h2, att_src2, att_dst2,
                                           asrc2, adst2, m2, s2, out2, N);
  edge_max_h1<<<cdivu(Et, B), B, 0, stream>>>(ei, E, N, asrc2, adst2, m2);
  edge_sum_h1<<<cdivu(Et, B), B, 0, stream>>>(ei, E, N, asrc2, adst2, m2, s2);
  edge_agg_h1<<<cdivu(Et * 4, B), B, 0, stream>>>(ei, E, N, asrc2, adst2, m2, h2, out2);
  finalize2<<<cdivu(N, B), B, 0, stream>>>(out2, s2, bias2, outp, N);
}
